// WAttention_32538672234925
// MI455X (gfx1250) — matrix operand order repaired and checked by reading, not compiled, not run on hardware
//
#include <hip/hip_runtime.h>
#include <hip/hip_bf16.h>

typedef __attribute__((ext_vector_type(16))) _Float16 v16h;
typedef __attribute__((ext_vector_type(8)))  float    v8f;
typedef __attribute__((ext_vector_type(4)))  float    f4;

#define WMMA(a, b, c) __builtin_amdgcn_wmma_f32_16x16x32_f16(false, (a), false, (b), (short)0, (c), false, false)

// ---------------------------------------------------------------------------
// CDNA5 async global->LDS copy path (ASYNCcnt-tracked), with safe fallback
// ---------------------------------------------------------------------------
#if defined(__HIP_DEVICE_COMPILE__) && __has_builtin(__builtin_amdgcn_global_load_async_to_lds_b128) && __has_builtin(__builtin_amdgcn_s_wait_asynccnt)
#define USE_ASYNC_LDS 1
#else
#define USE_ASYNC_LDS 0
#endif

#if USE_ASYNC_LDS
typedef int v4i_ __attribute__((vector_size(16)));
typedef __attribute__((address_space(1))) v4i_ gv4i;
typedef __attribute__((address_space(3))) v4i_ lv4i;
#endif

__device__ __forceinline__ void cp_b128(const _Float16* g, _Float16* l) {
#if USE_ASYNC_LDS
  __builtin_amdgcn_global_load_async_to_lds_b128((gv4i*)g, (lv4i*)l, 0, 0);
#else
  *(f4*)l = *(const f4*)g;
#endif
}
__device__ __forceinline__ void cp_fence() {
#if USE_ASYNC_LDS
  __builtin_amdgcn_s_wait_asynccnt(0);
#endif
}

// ---------------------------------------------------------------------------
// small packing kernels
// ---------------------------------------------------------------------------
__global__ __launch_bounds__(256) void pack_w3(const float* __restrict__ src, _Float16* __restrict__ dst) {
  int i = blockIdx.x * 256 + threadIdx.x;
  if (i >= 128 * 128 * 9) return;
  int kw = i % 3, r = i / 3;
  int kh = r % 3; r /= 3;
  int ci = r % 128; int o = r / 128;
  dst[o * 1152 + (kh * 3 + kw) * 128 + ci] = (_Float16)src[i];
}

__global__ __launch_bounds__(256) void pack_w1(const float* __restrict__ src, _Float16* __restrict__ dst, int n) {
  int i = blockIdx.x * 256 + threadIdx.x;
  if (i < n) dst[i] = (_Float16)src[i];
}

// x NCHW f32 -> padded NHWC f16 (pad=1, zeros in halo)
__global__ __launch_bounds__(256) void pack_pad_x(const float* __restrict__ x, _Float16* __restrict__ xp) {
  size_t i = (size_t)blockIdx.x * 256 + threadIdx.x;
  if (i >= (size_t)16 * 130 * 130 * 128) return;
  int c = (int)(i & 127);
  size_t r = i >> 7;
  int wp = (int)(r % 130); r /= 130;
  int hp = (int)(r % 130); int b = (int)(r / 130);
  float v = 0.f;
  if (hp >= 1 && hp <= 128 && wp >= 1 && wp <= 128)
    v = x[(((size_t)(b * 128 + c)) * 128 + (hp - 1)) * 128 + (wp - 1)];
  xp[i] = (_Float16)v;
}

// bilinear align_corners upsample of y (64->128) fused with token LayerNorm -> NHWC f16
__global__ __launch_bounds__(128) void upsample_ln_y(const float* __restrict__ y,
                                                     const float* __restrict__ lw,
                                                     const float* __restrict__ lb,
                                                     _Float16* __restrict__ out) {
  __shared__ float sv[128], sq[128];
  __shared__ float smu, srs;
  int p = blockIdx.x;
  int b = p >> 14, rem = p & 16383, h = rem >> 7, w = rem & 127;
  int c = threadIdx.x;
  float ry = h * (63.f / 127.f);
  int y0 = (int)ry; float wy = ry - y0; int y1 = y0 + 1; if (y1 > 63) y1 = 63;
  float rx = w * (63.f / 127.f);
  int x0 = (int)rx; float wx = rx - x0; int x1 = x0 + 1; if (x1 > 63) x1 = 63;
  size_t base = ((size_t)(b * 128 + c)) * 4096;
  float v00 = y[base + y0 * 64 + x0], v01 = y[base + y0 * 64 + x1];
  float v10 = y[base + y1 * 64 + x0], v11 = y[base + y1 * 64 + x1];
  float v = (v00 * (1.f - wx) + v01 * wx) * (1.f - wy) + (v10 * (1.f - wx) + v11 * wx) * wy;
  sv[c] = v; sq[c] = v * v;
  __syncthreads();
  if (c == 0) {
    float s = 0.f, s2 = 0.f;
    for (int i = 0; i < 128; ++i) { s += sv[i]; s2 += sq[i]; }
    float m = s * (1.f / 128.f);
    float var = s2 * (1.f / 128.f) - m * m;
    smu = m; srs = rsqrtf(var + 1e-5f);
  }
  __syncthreads();
  out[((size_t)p) * 128 + c] = (_Float16)((v - smu) * srs * lw[c] + lb[c]);
}

// ---------------------------------------------------------------------------
// fragment helpers (wave32 WMMA f16 layouts)
// ---------------------------------------------------------------------------
__device__ __forceinline__ v16h fragA(const _Float16* p /* &A[row][colbase+kb] */) {
  v16h a;
  *(f4*)&a = *(const f4*)p;
  *(((f4*)&a) + 1) = *(const f4*)(p + 16);
  return a;
}
__device__ __forceinline__ v16h fragB(const _Float16* p /* &Bt[n][kb] */) {
  v16h b;
  *(f4*)&b = *(const f4*)p;
  *(((f4*)&b) + 1) = *(const f4*)(p + 16);
  return b;
}

// ---------------------------------------------------------------------------
// conv 3x3 implicit GEMM: M=64 px, N=128 Cout, K=1152; fused LN_cf(1e-6)+relu
// double-buffered async weight staging
// ---------------------------------------------------------------------------
__global__ __launch_bounds__(256) void conv3x3_ln_relu(const _Float16* __restrict__ inpad,
                                                       const _Float16* __restrict__ wt,
                                                       const float* __restrict__ cb,
                                                       const float* __restrict__ lnw,
                                                       const float* __restrict__ lnb,
                                                       _Float16* __restrict__ out) {
  __shared__ _Float16 Bst[2][128][32];
  __shared__ float Cs[64][128];
  __shared__ float mu[64], rs[64];
  int blk = blockIdx.x;
  int tile = blk & 1, hrow = (blk >> 1) & 127, b = blk >> 8;
  int w0 = tile * 64;
  int t = threadIdx.x, wave = t >> 5, lane = t & 31;
  int mT = wave >> 1, nBase = (wave & 1) * 64;
  int lrow = lane & 15, hiHalf = lane >> 4;
  int sn = t >> 1, sseg = t & 1;
  v8f acc[4] = {};

  {  // stage chunk 0
    const _Float16* src = wt + sn * 1152 + sseg * 16;
    _Float16* dst = &Bst[0][sn][sseg * 16];
    cp_b128(src, dst); cp_b128(src + 8, dst + 8);
  }
  cp_fence();
  __syncthreads();

  for (int ck = 0; ck < 36; ++ck) {
    if (ck + 1 < 36) {  // stage next chunk into the other buffer
      const _Float16* src = wt + sn * 1152 + (ck + 1) * 32 + sseg * 16;
      _Float16* dst = &Bst[(ck + 1) & 1][sn][sseg * 16];
      cp_b128(src, dst); cp_b128(src + 8, dst + 8);
    }
    int pos = ck >> 2, cbase = (ck & 3) * 32;
    int kh = pos / 3, kw = pos - kh * 3;
    int px = mT * 16 + lrow;
    const _Float16* ap = inpad + (((size_t)(b * 130 + hrow + kh) * 130) + (w0 + px + kw)) * 128
                               + cbase + hiHalf * 8;
    __builtin_prefetch(ap + 32, 0, 1);
    v16h a = fragA(ap);
    int kbB = hiHalf * 8;
    const _Float16 (*B)[32] = Bst[ck & 1];
    for (int j = 0; j < 4; ++j) {
      v16h bf = fragB(&B[nBase + j * 16 + lrow][kbB]);
      acc[j] = WMMA(a, bf, acc[j]);
    }
    cp_fence();
    __syncthreads();
  }
  {
    int m0 = mT * 16 + hiHalf * 8;
    for (int j = 0; j < 4; ++j) {
      int n = nBase + j * 16 + lrow;
      float bb = cb[n];
      for (int r = 0; r < 8; ++r) Cs[m0 + r][n] = acc[j][r] + bb;
    }
  }
  __syncthreads();
  if (t < 64) {
    float s = 0.f, s2 = 0.f;
    for (int c = 0; c < 128; ++c) { float v = Cs[t][c]; s += v; s2 += v * v; }
    float m = s * (1.f / 128.f);
    float var = s2 * (1.f / 128.f) - m * m;
    mu[t] = m; rs[t] = rsqrtf(var + 1e-6f);
  }
  __syncthreads();
  for (int e = t; e < 64 * 128; e += 256) {
    int px = e >> 7, c = e & 127;
    float v = (Cs[px][c] - mu[px]) * rs[px] * lnw[c] + lnb[c];
    v = v > 0.f ? v : 0.f;
    out[(((size_t)(b * 128 + hrow) * 128) + (w0 + px)) * 128 + c] = (_Float16)v;
  }
}

// ---------------------------------------------------------------------------
// conv 1x1 GEMM: M=64 px, N=128, K=128. mode 0: LN_cf+relu -> f16 NHWC
// mode 1: LN_cf + residual(padded f16) + relu -> f32 NCHW (final output)
// ---------------------------------------------------------------------------
__global__ __launch_bounds__(256) void conv1x1_ln(const _Float16* __restrict__ in,
                                                  const _Float16* __restrict__ wt,
                                                  const float* __restrict__ cb,
                                                  const float* __restrict__ lnw,
                                                  const float* __restrict__ lnb,
                                                  const _Float16* __restrict__ residpad,
                                                  _Float16* __restrict__ outh,
                                                  float* __restrict__ outf,
                                                  int mode) {
  __shared__ _Float16 Bst[2][128][32];
  __shared__ float Cs[64][128];
  __shared__ float mu[64], rs[64];
  int blk = blockIdx.x;
  int tile = blk & 1, hrow = (blk >> 1) & 127, b = blk >> 8;
  int w0 = tile * 64;
  int t = threadIdx.x, wave = t >> 5, lane = t & 31;
  int mT = wave >> 1, nBase = (wave & 1) * 64;
  int lrow = lane & 15, hiHalf = lane >> 4;
  int sn = t >> 1, sseg = t & 1;
  v8f acc[4] = {};

  {
    const _Float16* src = wt + sn * 128 + sseg * 16;
    _Float16* dst = &Bst[0][sn][sseg * 16];
    cp_b128(src, dst); cp_b128(src + 8, dst + 8);
  }
  cp_fence();
  __syncthreads();

  for (int ck = 0; ck < 4; ++ck) {
    if (ck + 1 < 4) {
      const _Float16* src = wt + sn * 128 + (ck + 1) * 32 + sseg * 16;
      _Float16* dst = &Bst[(ck + 1) & 1][sn][sseg * 16];
      cp_b128(src, dst); cp_b128(src + 8, dst + 8);
    }
    int px = mT * 16 + lrow;
    const _Float16* ap = in + (((size_t)(b * 128 + hrow) * 128) + (w0 + px)) * 128
                            + ck * 32 + hiHalf * 8;
    v16h a = fragA(ap);
    int kbB = hiHalf * 8;
    const _Float16 (*B)[32] = Bst[ck & 1];
    for (int j = 0; j < 4; ++j) {
      v16h bf = fragB(&B[nBase + j * 16 + lrow][kbB]);
      acc[j] = WMMA(a, bf, acc[j]);
    }
    cp_fence();
    __syncthreads();
  }
  {
    int m0 = mT * 16 + hiHalf * 8;
    for (int j = 0; j < 4; ++j) {
      int n = nBase + j * 16 + lrow;
      float bb = cb[n];
      for (int r = 0; r < 8; ++r) Cs[m0 + r][n] = acc[j][r] + bb;
    }
  }
  __syncthreads();
  if (t < 64) {
    float s = 0.f, s2 = 0.f;
    for (int c = 0; c < 128; ++c) { float v = Cs[t][c]; s += v; s2 += v * v; }
    float m = s * (1.f / 128.f);
    float var = s2 * (1.f / 128.f) - m * m;
    mu[t] = m; rs[t] = rsqrtf(var + 1e-6f);
  }
  __syncthreads();
  for (int e = t; e < 64 * 128; e += 256) {
    int px = e >> 7, c = e & 127;
    float v = (Cs[px][c] - mu[px]) * rs[px] * lnw[c] + lnb[c];
    if (mode == 0) {
      v = v > 0.f ? v : 0.f;
      outh[(((size_t)(b * 128 + hrow) * 128) + (w0 + px)) * 128 + c] = (_Float16)v;
    } else {
      float res = (float)residpad[(((size_t)(b * 130 + hrow + 1) * 130) + (w0 + px + 1)) * 128 + c];
      v += res;
      v = v > 0.f ? v : 0.f;
      outf[(((size_t)(b * 128 + c) * 128) + hrow) * 128 + (w0 + px)] = v;
    }
  }
}

// ---------------------------------------------------------------------------
// token GEMM helpers: Out[64][128] = A[64][128] @ Bt^T + bias (Bt is [n][k])
// ---------------------------------------------------------------------------
__device__ __forceinline__ void gemm_tok(const _Float16 (*A)[128], const _Float16* __restrict__ Bt,
                                         const float* __restrict__ bias,
                                         _Float16 (*Bst)[128][32], _Float16 (*Out)[128], int t) {
  int wave = t >> 5, lane = t & 31;
  int mT = wave >> 1, nBase = (wave & 1) * 64;
  int lrow = lane & 15, hiHalf = lane >> 4;
  int sn = t >> 1, sseg = t & 1;
  v8f acc[4] = {};
  {
    const _Float16* src = Bt + sn * 128 + sseg * 16;
    _Float16* dst = &Bst[0][sn][sseg * 16];
    cp_b128(src, dst); cp_b128(src + 8, dst + 8);
  }
  cp_fence();
  __syncthreads();
  for (int ck = 0; ck < 4; ++ck) {
    if (ck + 1 < 4) {
      const _Float16* src = Bt + sn * 128 + (ck + 1) * 32 + sseg * 16;
      _Float16* dst = &Bst[(ck + 1) & 1][sn][sseg * 16];
      cp_b128(src, dst); cp_b128(src + 8, dst + 8);
    }
    v16h a = fragA(&A[mT * 16 + lrow][ck * 32 + hiHalf * 8]);
    int kbB = hiHalf * 8;
    const _Float16 (*B)[32] = Bst[ck & 1];
    for (int j = 0; j < 4; ++j) {
      v16h bf = fragB(&B[nBase + j * 16 + lrow][kbB]);
      acc[j] = WMMA(a, bf, acc[j]);
    }
    cp_fence();
    __syncthreads();
  }
  int m0 = mT * 16 + hiHalf * 8;
  for (int j = 0; j < 4; ++j) {
    int n = nBase + j * 16 + lrow;
    float bb = bias[n];
    for (int r = 0; r < 8; ++r) Out[m0 + r][n] = (_Float16)(acc[j][r] + bb);
  }
}

__device__ __forceinline__ void gemm_tok_f32(const _Float16 (*A)[128], const _Float16* __restrict__ Bt,
                                             const float* __restrict__ bias,
                                             _Float16 (*Bst)[128][32], float (*Out)[128], int t) {
  int wave = t >> 5, lane = t & 31;
  int mT = wave >> 1, nBase = (wave & 1) * 64;
  int lrow = lane & 15, hiHalf = lane >> 4;
  int sn = t >> 1, sseg = t & 1;
  v8f acc[4] = {};
  {
    const _Float16* src = Bt + sn * 128 + sseg * 16;
    _Float16* dst = &Bst[0][sn][sseg * 16];
    cp_b128(src, dst); cp_b128(src + 8, dst + 8);
  }
  cp_fence();
  __syncthreads();
  for (int ck = 0; ck < 4; ++ck) {
    if (ck + 1 < 4) {
      const _Float16* src = Bt + sn * 128 + (ck + 1) * 32 + sseg * 16;
      _Float16* dst = &Bst[(ck + 1) & 1][sn][sseg * 16];
      cp_b128(src, dst); cp_b128(src + 8, dst + 8);
    }
    v16h a = fragA(&A[mT * 16 + lrow][ck * 32 + hiHalf * 8]);
    int kbB = hiHalf * 8;
    const _Float16 (*B)[32] = Bst[ck & 1];
    for (int j = 0; j < 4; ++j) {
      v16h bf = fragB(&B[nBase + j * 16 + lrow][kbB]);
      acc[j] = WMMA(a, bf, acc[j]);
    }
    cp_fence();
    __syncthreads();
  }
  int m0 = mT * 16 + hiHalf * 8;
  for (int j = 0; j < 4; ++j) {
    int n = nBase + j * 16 + lrow;
    float bb = bias[n];
    for (int r = 0; r < 8; ++r) Out[m0 + r][n] = acc[j][r] + bb;
  }
}

// ---------------------------------------------------------------------------
// windowed cross-attention: one workgroup per 8x8 window (4096 windows)
// ---------------------------------------------------------------------------
__global__ __launch_bounds__(256) void win_attn(const _Float16* __restrict__ xtok,
                                                const _Float16* __restrict__ ytok,
                                                const _Float16* __restrict__ qw,
                                                const float* __restrict__ qb,
                                                const _Float16* __restrict__ kvw,
                                                const float* __restrict__ kvb,
                                                const _Float16* __restrict__ mlpw,
                                                const float* __restrict__ mlpb,
                                                const float* __restrict__ lnxw,
                                                const float* __restrict__ lnxb,
                                                const float* __restrict__ mlw,
                                                const float* __restrict__ mlb,
                                                _Float16* __restrict__ outpad) {
  __shared__ _Float16 Xs[64][128];
  __shared__ _Float16 Qs[64][128];
  __shared__ _Float16 Ks[64][128];
  __shared__ _Float16 Vs[64][128];
  __shared__ float    Ss[64][64];
  __shared__ _Float16 Ps[64][64];
  __shared__ _Float16 Bst[2][128][32];
  __shared__ float    mu[64], rs[64];
  __shared__ float    Scr[64][128];  // overlay: {XL | Ys} then Os
  _Float16 (*XL)[128] = (_Float16 (*)[128]) & Scr[0][0];
  _Float16 (*Ys)[128] = (_Float16 (*)[128]) & Scr[32][0];
  float (*Os)[128] = Scr;

  int blk = blockIdx.x;
  int b = blk >> 8, wi = blk & 255, wh = wi >> 4, ww = wi & 15;
  int t = threadIdx.x, wave = t >> 5, lane = t & 31;
  int lrow = lane & 15, hiHalf = lane >> 4;

  // 1. load tokens (async to LDS)
  for (int f = t; f < 1024; f += 256) {
    int tok = f >> 4, seg = f & 15;
    int ti = tok >> 3, tj = tok & 7;
    size_t base = (((size_t)(b * 128 + wh * 8 + ti) * 128) + (ww * 8 + tj)) * 128 + seg * 8;
    cp_b128(xtok + base, &Xs[tok][seg * 8]);
    cp_b128(ytok + base, &Ys[tok][seg * 8]);
  }
  cp_fence();
  __syncthreads();

  // 2. LN(x tokens) -> XL (eps 1e-5)
  if (t < 64) {
    float s = 0.f, s2 = 0.f;
    for (int c = 0; c < 128; ++c) { float v = (float)Xs[t][c]; s += v; s2 += v * v; }
    float m = s * (1.f / 128.f);
    float var = s2 * (1.f / 128.f) - m * m;
    mu[t] = m; rs[t] = rsqrtf(var + 1e-5f);
  }
  __syncthreads();
  for (int e = t; e < 8192; e += 256) {
    int tok = e >> 7, c = e & 127;
    XL[tok][c] = (_Float16)(((float)Xs[tok][c] - mu[tok]) * rs[tok] * lnxw[c] + lnxb[c]);
  }
  __syncthreads();

  // 3. projections
  gemm_tok(XL, qw, qb, Bst, Qs, t);
  __syncthreads();
  gemm_tok(Ys, kvw, kvb, Bst, Ks, t);
  __syncthreads();
  gemm_tok(Ys, kvw + 128 * 128, kvb + 128, Bst, Vs, t);
  __syncthreads();

  // 4. attention per head (hd=16, zero-padded to K=32)
  for (int hd = 0; hd < 8; ++hd) {
    for (int tt = 0; tt < 2; ++tt) {
      int T = wave + tt * 8;
      int mT = T >> 2, nT = T & 3;
      v16h a = {};
      *(f4*)&a = *(const f4*)&Qs[mT * 16 + lrow][hd * 16 + hiHalf * 8];  // upper K half stays 0
      v16h bf = {};
      {
        const _Float16* bp = &Ks[nT * 16 + lrow][hd * 16 + hiHalf * 8];
        *(f4*)&bf = *(const f4*)bp;
      }
      v8f c = {};
      c = WMMA(a, bf, c);
      int n = nT * 16 + lrow, m0 = mT * 16 + hiHalf * 8;
      for (int r = 0; r < 8; ++r) Ss[m0 + r][n] = c[r] * 0.25f;  // scale = hd^-0.5
    }
    __syncthreads();
    if (t < 64) {
      float mx = -1e30f;
      for (int j = 0; j < 64; ++j) mx = fmaxf(mx, Ss[t][j]);
      float sum = 0.f;
      for (int j = 0; j < 64; ++j) { float e = __expf(Ss[t][j] - mx); Ss[t][j] = e; sum += e; }
      float inv = 1.f / sum;
      for (int j = 0; j < 64; ++j) Ps[t][j] = (_Float16)(Ss[t][j] * inv);
    }
    __syncthreads();
    if (wave < 4) {
      int mT = wave;
      v8f c = {};
      for (int ck = 0; ck < 2; ++ck) {
        v16h a = fragA(&Ps[mT * 16 + lrow][ck * 32 + hiHalf * 8]);
        v16h bf;
        int kb = hiHalf * 8;
        for (int j = 0; j < 16; ++j) bf[j] = Vs[ck * 32 + kb + j + (j >= 8 ? 8 : 0)][hd * 16 + lrow];  // V^T gather
        c = WMMA(a, bf, c);
      }
      int m0 = mT * 16 + hiHalf * 8;
      for (int r = 0; r < 8; ++r) Os[m0 + r][hd * 16 + lrow] = c[r];
    }
    __syncthreads();
  }

  // 5. xt = relu(attn_out + sc1)
  for (int e = t; e < 8192; e += 256) {
    int tok = e >> 7, c = e & 127;
    float v = Os[tok][c] + (float)Xs[tok][c];
    Xs[tok][c] = (_Float16)(v > 0.f ? v : 0.f);
  }
  __syncthreads();

  // 6. MLP linear -> Os (f32)
  gemm_tok_f32(Xs, mlpw, mlpb, Bst, Os, t);
  __syncthreads();

  // 7. LN of MLP output (eps 1e-5)
  if (t < 64) {
    float s = 0.f, s2 = 0.f;
    for (int c = 0; c < 128; ++c) { float v = Os[t][c]; s += v; s2 += v * v; }
    float m = s * (1.f / 128.f);
    float var = s2 * (1.f / 128.f) - m * m;
    mu[t] = m; rs[t] = rsqrtf(var + 1e-5f);
  }
  __syncthreads();

  // 8. mlp = relu(LN + xt); out = relu(xt + mlp); scatter into padded NHWC
  for (int e = t; e < 8192; e += 256) {
    int tok = e >> 7, c = e & 127;
    float xt = (float)Xs[tok][c];
    float ln = (Os[tok][c] - mu[tok]) * rs[tok] * mlw[c] + mlb[c];
    float mlp = ln + xt; mlp = mlp > 0.f ? mlp : 0.f;
    float res = xt + mlp; res = res > 0.f ? res : 0.f;
    int ti = tok >> 3, tj = tok & 7;
    size_t o = (((size_t)(b * 130 + wh * 8 + ti + 1) * 130) + (ww * 8 + tj + 1)) * 128 + c;
    outpad[o] = (_Float16)res;
  }
}

// ---------------------------------------------------------------------------
// launch
// ---------------------------------------------------------------------------
extern "C" void kernel_launch(void* const* d_in, const int* in_sizes, int n_in,
                              void* d_out, int out_size, void* d_ws, size_t ws_size,
                              hipStream_t stream) {
  const float* x = (const float*)d_in[0];
  const float* y = (const float*)d_in[1];
  const float* convx1_w = (const float*)d_in[2];
  const float* convx1_b = (const float*)d_in[3];
  const float* lncx1_w  = (const float*)d_in[4];
  const float* lncx1_b  = (const float*)d_in[5];
  const float* convx2_w = (const float*)d_in[6];
  const float* convx2_b = (const float*)d_in[7];
  const float* lncx2_w  = (const float*)d_in[8];
  const float* lncx2_b  = (const float*)d_in[9];
  const float* lnx_w    = (const float*)d_in[10];
  const float* lnx_b    = (const float*)d_in[11];
  const float* lny_w    = (const float*)d_in[12];
  const float* lny_b    = (const float*)d_in[13];
  const float* q_w      = (const float*)d_in[14];
  const float* q_b      = (const float*)d_in[15];
  const float* kv_w     = (const float*)d_in[16];
  const float* kv_b     = (const float*)d_in[17];
  const float* mlp_w    = (const float*)d_in[18];
  const float* mlp_b    = (const float*)d_in[19];
  const float* mlpln_w  = (const float*)d_in[20];
  const float* mlpln_b  = (const float*)d_in[21];
  const float* c2a_w    = (const float*)d_in[22];
  const float* c2a_b    = (const float*)d_in[23];
  const float* c2ln1_w  = (const float*)d_in[24];
  const float* c2ln1_b  = (const float*)d_in[25];
  const float* c2b_w    = (const float*)d_in[26];
  const float* c2b_b    = (const float*)d_in[27];
  const float* c2ln2_w  = (const float*)d_in[28];
  const float* c2ln2_b  = (const float*)d_in[29];

  const size_t ELEMS = (size_t)16 * 128 * 128 * 128;      // 33.5M
  const size_t PAD_ELEMS = (size_t)16 * 130 * 130 * 128;  // 34.6M

  char* ws = (char*)d_ws;
  size_t off = 0;
  auto carve = [&](size_t bytes) { char* p = ws + off; off += (bytes + 255) & ~(size_t)255; return p; };

  _Float16* xpad = (_Float16*)carve(PAD_ELEMS * 2);  // padded NHWC input x
  _Float16* spad = (_Float16*)carve(PAD_ELEMS * 2);  // padded NHWC attention output (xs)
  _Float16* s1   = (_Float16*)carve(ELEMS * 2);      // stem conv1 out; reused for conv2a out
  _Float16* x2   = (_Float16*)carve(ELEMS * 2);      // stem out (tokens)
  _Float16* yln  = (_Float16*)carve(ELEMS * 2);      // upsampled + LN'd y
  _Float16* w3a  = (_Float16*)carve((size_t)128 * 1152 * 2);
  _Float16* w3b  = (_Float16*)carve((size_t)128 * 1152 * 2);
  _Float16* w1a  = (_Float16*)carve((size_t)128 * 128 * 2);
  _Float16* qwh  = (_Float16*)carve((size_t)128 * 128 * 2);
  _Float16* kvwh = (_Float16*)carve((size_t)256 * 128 * 2);
  _Float16* mlwh = (_Float16*)carve((size_t)128 * 128 * 2);
  _Float16* w1c  = (_Float16*)carve((size_t)128 * 128 * 2);
  (void)ws_size; (void)in_sizes; (void)n_in; (void)out_size;

  // weight packing (f16, Bt=[N][K] layout)
  pack_w3<<<(128 * 128 * 9 + 255) / 256, 256, 0, stream>>>(convx1_w, w3a);
  pack_w3<<<(128 * 128 * 9 + 255) / 256, 256, 0, stream>>>(c2a_w, w3b);
  pack_w1<<<(16384 + 255) / 256, 256, 0, stream>>>(convx2_w, w1a, 16384);
  pack_w1<<<(16384 + 255) / 256, 256, 0, stream>>>(q_w, qwh, 16384);
  pack_w1<<<(32768 + 255) / 256, 256, 0, stream>>>(kv_w, kvwh, 32768);
  pack_w1<<<(16384 + 255) / 256, 256, 0, stream>>>(mlp_w, mlwh, 16384);
  pack_w1<<<(16384 + 255) / 256, 256, 0, stream>>>(c2b_w, w1c, 16384);

  // activation packing
  pack_pad_x<<<(unsigned)((PAD_ELEMS + 255) / 256), 256, 0, stream>>>(x, xpad);
  upsample_ln_y<<<16 * 128 * 128, 128, 0, stream>>>(y, lny_w, lny_b, yln);
  (void)hipMemsetAsync(spad, 0, PAD_ELEMS * 2, stream);  // halo zeros for conv2a

  // stem: conv3x3+LN+relu -> conv1x1+LN+relu
  conv3x3_ln_relu<<<4096, 256, 0, stream>>>(xpad, w3a, convx1_b, lncx1_w, lncx1_b, s1);
  conv1x1_ln<<<4096, 256, 0, stream>>>(s1, w1a, convx2_b, lncx2_w, lncx2_b, nullptr, x2, nullptr, 0);

  // windowed cross attention + MLP
  win_attn<<<4096, 256, 0, stream>>>(x2, yln, qwh, q_b, kvwh, kv_b, mlwh, mlp_b,
                                     lnx_w, lnx_b, mlpln_w, mlpln_b, spad);

  // conv2: conv3x3+LN+relu -> conv1x1+LN + residual + relu -> NCHW f32 out
  conv3x3_ln_relu<<<4096, 256, 0, stream>>>(spad, w3b, c2a_b, c2ln1_w, c2ln1_b, s1);
  conv1x1_ln<<<4096, 256, 0, stream>>>(s1, w1c, c2b_b, c2ln2_w, c2ln2_b, spad, nullptr,
                                       (float*)d_out, 1);
}
